// RGCN_DualAttn_FFNN_3985729651462
// MI455X (gfx1250) — matrix operand order repaired and checked by reading, not compiled, not run on hardware
//
#include <hip/hip_runtime.h>
#include <hip/hip_bf16.h>

#define DIM     128
#define NREL    6
#define NHEADS  8
#define SLEN    16
#define FC1     192   // 3*DIM/2

typedef __attribute__((ext_vector_type(16))) __bf16 v16bf;
typedef __attribute__((ext_vector_type(8)))  __bf16 v8bf;
typedef __attribute__((ext_vector_type(8)))  float  v8f;

// round-to-nearest-even f32 -> bf16 (used only in bandwidth-bound passes)
__device__ __forceinline__ __bf16 f2bf(float f) {
    union { float f; unsigned u; } cv; cv.f = f;
    unsigned u = cv.u;
    u += 0x7FFFu + ((u >> 16) & 1u);
    union { unsigned short s; __bf16 b; } cb;
    cb.s = (unsigned short)(u >> 16);
    return cb.b;
}
__device__ __forceinline__ float bf2f(unsigned short s) {
    union { unsigned u; float f; } c; c.u = ((unsigned)s) << 16; return c.f;
}

// ---------------------------------------------------------------- utilities
__global__ void fill_zero_f32(float* __restrict__ p, long long n) {
    long long i = (long long)blockIdx.x * blockDim.x + threadIdx.x;
    long long st = (long long)gridDim.x * blockDim.x;
    for (; i < n; i += st) p[i] = 0.0f;
}
__global__ void fill_zero_i32(int* __restrict__ p, long long n) {
    long long i = (long long)blockIdx.x * blockDim.x + threadIdx.x;
    long long st = (long long)gridDim.x * blockDim.x;
    for (; i < n; i += st) p[i] = 0;
}

// packed f32 -> bf16 conversion, 4 elements/thread (n must be multiple of 4)
__global__ void cvt_f32_bf16(const float* __restrict__ in, __bf16* __restrict__ out,
                             long long n4) {
    long long i = (long long)blockIdx.x * blockDim.x + threadIdx.x;
    long long st = (long long)gridDim.x * blockDim.x;
    for (; i < n4; i += st) {
        const float4 v = ((const float4*)in)[i];
        union { __bf16 b[4]; uint2 u; } o;
        o.b[0] = f2bf(v.x); o.b[1] = f2bf(v.y);
        o.b[2] = f2bf(v.z); o.b[3] = f2bf(v.w);
        ((uint2*)out)[i] = o.u;
    }
}

// weights: [6][k][n] + root[k][n] f32  ->  wT[7][n][k] bf16 (B-fragment friendly)
__global__ void make_wT(const float* __restrict__ wrel, const float* __restrict__ wroot,
                        __bf16* __restrict__ wT) {
    int idx = blockIdx.x * blockDim.x + threadIdx.x;
    if (idx >= (NREL + 1) * DIM * DIM) return;
    const int mat = idx / (DIM * DIM);
    const int rem = idx - mat * DIM * DIM;
    const int n = rem / DIM, k = rem % DIM;
    const float* src = (mat < NREL) ? (wrel + (long long)mat * DIM * DIM) : wroot;
    wT[idx] = f2bf(src[k * DIM + n]);
}

// ------------------------------------------------- per-(relation,dst) degree
__global__ void count_deg(const int* __restrict__ dst, const int* __restrict__ etype,
                          int* __restrict__ cnt, int E, int N) {
    int e = blockIdx.x * blockDim.x + threadIdx.x;
    if (e < E) atomicAdd(&cnt[(long long)etype[e] * N + dst[e]], 1);
}

// --------------------------- scatter xb[src]*norm into agg[r,dst,:] (f32 acc)
// one wave per edge; lane moves 4 bf16 (8B gather) -> 4 f32 atomics
__global__ void __launch_bounds__(256)
scatter_edges(const __bf16* __restrict__ xb, const int* __restrict__ src,
              const int* __restrict__ dst, const int* __restrict__ etype,
              const int* __restrict__ cnt, float* __restrict__ agg,
              int E, int N) {
    int e = blockIdx.x * blockDim.y + threadIdx.y;
    if (e >= E) return;
    const int lane = threadIdx.x;
    const int s = src[e], d = dst[e], t = etype[e];
    const float nrm = 1.0f / fmaxf((float)cnt[(long long)t * N + d], 1.0f);
    union { uint2 u; unsigned short s4[4]; } cv;
    cv.u = ((const uint2*)(xb + (long long)s * DIM))[lane];
    float* ar = agg + ((long long)t * N + d) * DIM + lane * 4;
    atomicAdd(ar + 0, bf2f(cv.s4[0]) * nrm);
    atomicAdd(ar + 1, bf2f(cv.s4[1]) * nrm);
    atomicAdd(ar + 2, bf2f(cv.s4[2]) * nrm);
    atomicAdd(ar + 3, bf2f(cv.s4[3]) * nrm);
}

// --------------- out = sum_r aggb[r] @ w_rel[r] + xb @ w_root + b  (bf16 WMMA)
// wave computes one 16x16 tile; hot loop = 4x global_load_b128 + 1 wmma
__global__ void __launch_bounds__(256)
rgcn_gemm(const __bf16* __restrict__ xb, const __bf16* __restrict__ aggb,
          const __bf16* __restrict__ wT,           // [7][n][k] bf16
          const float* __restrict__ bias,
          float* __restrict__ out32, __bf16* __restrict__ out16,
          int N, int relu) {
    const int lane = threadIdx.x;      // 0..31
    const int wv   = threadIdx.y;      // 0..7 -> one 16-col tile each
    const int m0   = blockIdx.x * 16;
    const int n0   = wv * 16;
    const bool hi  = lane >= 16;
    const int  r   = lane & 15;
    int mrow = m0 + r; if (mrow >= N) mrow = N - 1;   // clamp (stores guarded)

    const int ka = hi ? 8 : 0;         // A: lanes<16 K=[0..7],[16..23]; hi +8
    const int kb = hi ? 8 : 0;         // B: lanes<16 rows K=[0..7],[16..23]; hi +8

    v8f acc = {0.f, 0.f, 0.f, 0.f, 0.f, 0.f, 0.f, 0.f};

    for (int rel = 0; rel < NREL + 1; ++rel) {
        const __bf16* arow =
            ((rel < NREL) ? (aggb + (long long)rel * N * DIM) : xb)
            + (long long)mrow * DIM;
        const __bf16* brow = wT + ((long long)rel * DIM + (n0 + r)) * DIM;
        #pragma unroll
        for (int k0 = 0; k0 < DIM; k0 += 32) {
            const v8bf alo = *(const v8bf*)(arow + k0 + ka);
            const v8bf ahi = *(const v8bf*)(arow + k0 + ka + 16);
            const v8bf blo = *(const v8bf*)(brow + k0 + kb);
            const v8bf bhi = *(const v8bf*)(brow + k0 + kb + 16);
            const v16bf a = __builtin_shufflevector(alo, ahi,
                0, 1, 2, 3, 4, 5, 6, 7, 8, 9, 10, 11, 12, 13, 14, 15);
            const v16bf b = __builtin_shufflevector(blo, bhi,
                0, 1, 2, 3, 4, 5, 6, 7, 8, 9, 10, 11, 12, 13, 14, 15);
            acc = __builtin_amdgcn_wmma_f32_16x16x32_bf16(
                false, a, false, b, (short)0, acc, false, false);
        }
    }

    const float bn = bias[n0 + r];
    #pragma unroll
    for (int j = 0; j < 8; ++j) {      // C/D: VGPR j -> row m0 + j (+8 for hi half)
        const int m = m0 + j + (hi ? 8 : 0);
        if (m < N) {
            float v = acc[j] + bn;
            if (relu) v = fmaxf(v, 0.0f);
            const long long o = (long long)m * DIM + n0 + r;
            if (out32) out32[o] = v;
            if (out16) out16[o] = f2bf(v);
        }
    }
}

// ------------------------------- single-query MHA, one block (128 thr)/bill
__global__ void __launch_bounds__(DIM)
mha_q1(const float* __restrict__ h, const int* __restrict__ qidx,
       const int* __restrict__ kvidx, const float* __restrict__ in_w,
       const float* __restrict__ in_b, const float* __restrict__ out_w,
       const float* __restrict__ out_b, const float* __restrict__ mask,
       float* __restrict__ out) {
    __shared__ float x0[DIM];
    __shared__ float xkv[SLEN][DIM];
    __shared__ float qh[DIM];
    __shared__ float kh[SLEN][DIM];
    __shared__ float vh[SLEN][DIM];
    __shared__ float attn[NHEADS][SLEN];
    __shared__ float oh[DIM];
    const int t = threadIdx.x;
    const int b = blockIdx.x;

    x0[t] = h[(long long)qidx[b] * DIM + t];
    for (int s = 0; s < SLEN; ++s)
        xkv[s][t] = h[(long long)kvidx[b * SLEN + s] * DIM + t];
    __syncthreads();

    {   // q projection
        float acc = in_b[t];
        const float* w = in_w + t * DIM;
        for (int d = 0; d < DIM; ++d) acc += x0[d] * w[d];
        qh[t] = acc;
    }
    for (int s = 0; s < SLEN; ++s) {   // k, v projections
        float ak = in_b[DIM + t], av = in_b[2 * DIM + t];
        const float* wk = in_w + (DIM + t) * DIM;
        const float* wv = in_w + (2 * DIM + t) * DIM;
        for (int d = 0; d < DIM; ++d) {
            const float xv = xkv[s][d];
            ak += xv * wk[d];
            av += xv * wv[d];
        }
        kh[s][t] = ak;
        vh[s][t] = av;
    }
    __syncthreads();

    const int dh = DIM / NHEADS;                  // 16
    {   // 128 threads = 8 heads x 16 positions
        const int hd = t >> 4, s = t & 15;
        float sc = 0.f;
        for (int d = 0; d < dh; ++d) sc += qh[hd * dh + d] * kh[s][hd * dh + d];
        attn[hd][s] = sc * 0.25f + mask[s];       // 1/sqrt(16)
    }
    __syncthreads();
    if (t < NHEADS) {                              // softmax over S
        float mx = -1e30f;
        for (int s = 0; s < SLEN; ++s) mx = fmaxf(mx, attn[t][s]);
        float sum = 0.f;
        for (int s = 0; s < SLEN; ++s) { float e = __expf(attn[t][s] - mx); attn[t][s] = e; sum += e; }
        const float inv = 1.0f / sum;
        for (int s = 0; s < SLEN; ++s) attn[t][s] *= inv;
    }
    __syncthreads();
    {   // attn @ V
        const int hd = t >> 4;
        float acc = 0.f;
        for (int s = 0; s < SLEN; ++s) acc += attn[hd][s] * vh[s][t];
        oh[t] = acc;
    }
    __syncthreads();
    {   // output projection
        float acc = out_b[t];
        const float* w = out_w + t * DIM;
        for (int d = 0; d < DIM; ++d) acc += oh[d] * w[d];
        out[(long long)b * DIM + t] = acc;
    }
}

// ------------------------------------ concat -> fc1+relu -> fc2 -> score[b]
__global__ void __launch_bounds__(FC1)
ffn_score(const float* __restrict__ left, const float* __restrict__ right,
          const float* __restrict__ h, const int* __restrict__ legis_idx,
          const float* __restrict__ fc1_w, const float* __restrict__ fc1_b,
          const float* __restrict__ fc2_w, const float* __restrict__ fc2_b,
          float* __restrict__ out) {
    __shared__ float xin[3 * DIM];
    __shared__ float y1[FC1];
    const int t = threadIdx.x;     // 0..191
    const int b = blockIdx.x;
    const long long lrow = (long long)legis_idx[b] * DIM;
    for (int i = t; i < 3 * DIM; i += FC1) {
        float v;
        if      (i < DIM)     v = left [(long long)b * DIM + i];
        else if (i < 2 * DIM) v = right[(long long)b * DIM + (i - DIM)];
        else                  v = h[lrow + (i - 2 * DIM)];
        xin[i] = v;
    }
    __syncthreads();
    {
        float acc = fc1_b[t];
        const float* w = fc1_w + t * (3 * DIM);
        for (int k = 0; k < 3 * DIM; ++k) acc += xin[k] * w[k];
        y1[t] = fmaxf(acc, 0.0f) * fc2_w[t];     // fold fc2 weight in
    }
    __syncthreads();
    if (t == 0) {
        float s = fc2_b[0];
        for (int k = 0; k < FC1; ++k) s += y1[k];
        out[b] = s;
    }
}

// ---------------------------------------------------------------- launcher
extern "C" void kernel_launch(void* const* d_in, const int* in_sizes, int n_in,
                              void* d_out, int out_size, void* d_ws, size_t ws_size,
                              hipStream_t stream) {
    const float* node_emb   = (const float*)d_in[0];
    const float* conv1_w    = (const float*)d_in[1];
    const float* conv1_root = (const float*)d_in[2];
    const float* conv1_b    = (const float*)d_in[3];
    const float* conv2_w    = (const float*)d_in[4];
    const float* conv2_root = (const float*)d_in[5];
    const float* conv2_b    = (const float*)d_in[6];
    const float* left_in_w  = (const float*)d_in[7];
    const float* left_in_b  = (const float*)d_in[8];
    const float* left_out_w = (const float*)d_in[9];
    const float* left_out_b = (const float*)d_in[10];
    const float* right_in_w = (const float*)d_in[11];
    const float* right_in_b = (const float*)d_in[12];
    const float* right_out_w= (const float*)d_in[13];
    const float* right_out_b= (const float*)d_in[14];
    const float* fc1_w      = (const float*)d_in[15];
    const float* fc1_b      = (const float*)d_in[16];
    const float* fc2_w      = (const float*)d_in[17];
    const float* fc2_b      = (const float*)d_in[18];
    const float* spon_mask  = (const float*)d_in[19];
    const float* subj_mask  = (const float*)d_in[20];
    const int*   edge_index = (const int*)d_in[21];
    const int*   edge_type  = (const int*)d_in[22];
    const int*   legis_idx  = (const int*)d_in[23];
    const int*   bill_idx   = (const int*)d_in[24];
    const int*   spon_idx   = (const int*)d_in[25];
    const int*   subj_idx   = (const int*)d_in[26];

    const int N = in_sizes[0] / DIM;
    const int E = in_sizes[22];
    const int B = in_sizes[23];
    const int* src = edge_index;
    const int* dst = edge_index + E;

    // workspace carve (256B aligned slabs)
    char* ws = (char*)d_ws;
    size_t off = 0;
    auto carve = [&](size_t bytes) -> void* {
        void* p = ws + off;
        off += (bytes + 255) & ~(size_t)255;
        return p;
    };
    int*    cnt   = (int*)   carve((size_t)NREL * N * sizeof(int));
    float*  agg   = (float*) carve((size_t)NREL * N * DIM * sizeof(float));
    __bf16* aggb  = (__bf16*)carve((size_t)NREL * N * DIM * sizeof(__bf16));
    __bf16* x0b   = (__bf16*)carve((size_t)N * DIM * sizeof(__bf16));
    __bf16* h1b   = (__bf16*)carve((size_t)N * DIM * sizeof(__bf16));
    float*  h2f   = (float*) carve((size_t)N * DIM * sizeof(float));
    __bf16* wT1   = (__bf16*)carve((size_t)(NREL + 1) * DIM * DIM * sizeof(__bf16));
    __bf16* wT2   = (__bf16*)carve((size_t)(NREL + 1) * DIM * DIM * sizeof(__bf16));
    float*  left  = (float*) carve((size_t)B * DIM * sizeof(float));
    float*  right = (float*) carve((size_t)B * DIM * sizeof(float));
    (void)ws_size; (void)n_in; (void)out_size;

    const long long aggn = (long long)NREL * N * DIM;
    const dim3 wgp(32, 8);                       // 8 waves / workgroup
    const int  mtiles = (N + 15) / 16;
    const int  wTn = (NREL + 1) * DIM * DIM;

    // setup: degrees, bf16 node features, transposed bf16 weights
    fill_zero_i32<<<1024, 256, 0, stream>>>(cnt, (long long)NREL * N);
    count_deg<<<(E + 255) / 256, 256, 0, stream>>>(dst, edge_type, cnt, E, N);
    cvt_f32_bf16<<<2048, 256, 0, stream>>>(node_emb, x0b, (long long)N * DIM / 4);
    make_wT<<<(wTn + 255) / 256, 256, 0, stream>>>(conv1_w, conv1_root, wT1);
    make_wT<<<(wTn + 255) / 256, 256, 0, stream>>>(conv2_w, conv2_root, wT2);

    // layer 1: agg <- scatter(x0b); h1b = relu(wmma-gemm)
    fill_zero_f32<<<8192, 256, 0, stream>>>(agg, aggn);
    scatter_edges<<<(E + 7) / 8, wgp, 0, stream>>>(x0b, src, dst, edge_type, cnt, agg, E, N);
    cvt_f32_bf16<<<8192, 256, 0, stream>>>(agg, aggb, aggn / 4);
    rgcn_gemm<<<mtiles, wgp, 0, stream>>>(x0b, aggb, wT1, conv1_b,
                                          (float*)nullptr, h1b, N, 1);

    // layer 2: agg <- scatter(h1b); h2f = wmma-gemm (f32 out for heads)
    fill_zero_f32<<<8192, 256, 0, stream>>>(agg, aggn);
    scatter_edges<<<(E + 7) / 8, wgp, 0, stream>>>(h1b, src, dst, edge_type, cnt, agg, E, N);
    cvt_f32_bf16<<<8192, 256, 0, stream>>>(agg, aggb, aggn / 4);
    rgcn_gemm<<<mtiles, wgp, 0, stream>>>(h1b, aggb, wT2, conv2_b,
                                          h2f, (__bf16*)nullptr, N, 0);

    // dual attention over sponsors / subjects
    mha_q1<<<B, DIM, 0, stream>>>(h2f, bill_idx, spon_idx, left_in_w, left_in_b,
                                  left_out_w, left_out_b, spon_mask, left);
    mha_q1<<<B, DIM, 0, stream>>>(h2f, bill_idx, subj_idx, right_in_w, right_in_b,
                                  right_out_w, right_out_b, subj_mask, right);

    // head
    ffn_score<<<B, FC1, 0, stream>>>(left, right, h2f, legis_idx,
                                     fc1_w, fc1_b, fc2_w, fc2_b, (float*)d_out);
}
